// AMLDetector_46351287058699
// MI455X (gfx1250) — matrix operand order repaired and checked by reading, not compiled, not run on hardware
//
#include <hip/hip_runtime.h>
#include <hip/hip_bf16.h>

#define N_NODES 6144
#define N_EDGES 98304
#define T_STEPS 32
#define D_INF   128
#define DMODEL  256
#define D_FF    256
#define N_HEADS 4
#define D_HEAD  64
#define OUT_DIM 2
#define LN_EPS  1e-5f

// attention staging
#define NBMAX 384   // max keys per timestep block (mean ~192, sd ~14 -> hugely safe)
#define KVP   72    // K/V LDS row pitch in halfs (36 banks -> conflict-free column gather)
#define PSP   40    // P-staging row pitch in halfs (16B-aligned rows)

typedef __attribute__((ext_vector_type(16))) _Float16 v16h;
typedef __attribute__((ext_vector_type(8)))  _Float16 v8h;
typedef __attribute__((ext_vector_type(8)))  float    v8f;

#define SHUF16(lo, hi) __builtin_shufflevector(lo, hi, 0,1,2,3,4,5,6,7,8,9,10,11,12,13,14,15)
#define WMMA(a, b, c) __builtin_amdgcn_wmma_f32_16x16x32_f16(false, a, false, b, (short)0, c, false, false)

// ---------------- elementwise helpers ----------------

__global__ void fill_zero(float* p, int n) {
  int i = blockIdx.x * blockDim.x + threadIdx.x;
  if (i < n) p[i] = 0.f;
}

__global__ void cvt_f32_f16(const float* __restrict__ s, _Float16* __restrict__ d, int n) {
  int i = blockIdx.x * blockDim.x + threadIdx.x;
  if (i < n) d[i] = (_Float16)s[i];
}

// Wcat[n][k] = k<128 ? lin_l_w[n][k] : lin_r_w[n][k-128]
__global__ void concat_weight(const float* __restrict__ lw, const float* __restrict__ rw,
                              _Float16* __restrict__ d) {
  int i = blockIdx.x * blockDim.x + threadIdx.x;
  if (i >= DMODEL * DMODEL) return;
  int n = i >> 8, k = i & 255;
  float v = (k < D_INF) ? lw[n * D_INF + k] : rw[n * D_INF + (k - D_INF)];
  d[i] = (_Float16)v;
}

// ---------------- SAGE aggregation ----------------

__global__ void edge_agg(const int* __restrict__ ei, const float* __restrict__ x,
                         float* __restrict__ agg, float* __restrict__ cnt) {
  int i = blockIdx.x * blockDim.x + threadIdx.x;
  if (i >= N_EDGES * D_INF) return;
  int e = i >> 7, c = i & 127;
  int src = ei[e];
  int dst = ei[N_EDGES + e];
  atomicAdd(&agg[(size_t)dst * D_INF + c], x[(size_t)src * D_INF + c]);
  if (c == 0) atomicAdd(&cnt[dst], 1.f);
}

__global__ void build_xa16(const float* __restrict__ agg, const float* __restrict__ cnt,
                           const float* __restrict__ x, _Float16* __restrict__ xa) {
  int i = blockIdx.x * blockDim.x + threadIdx.x;
  if (i >= N_NODES * DMODEL) return;
  int n = i >> 8, k = i & 255;
  float v = (k < D_INF) ? agg[(size_t)n * D_INF + k] / fmaxf(cnt[n], 1.f)
                        : x[(size_t)n * D_INF + (k - D_INF)];
  xa[i] = (_Float16)v;
}

__global__ void block_starts(const int* __restrict__ ts, int* __restrict__ bs) {
  int t = blockIdx.x * blockDim.x + threadIdx.x;
  if (t > T_STEPS) return;
  int lo = 0, hi = N_NODES;
  while (lo < hi) { int mid = (lo + hi) >> 1; if (ts[mid] < t) lo = mid + 1; else hi = mid; }
  bs[t] = lo;
}

// ---------------- WMMA GEMM: C[M x Nc] = A[M x K](f16) @ W[Nc x K]^T + bias ----------------
// One wave computes a 16x64 output panel (4 n-tiles share the A fragment).
// All 10 fragment loads of a k-step are issued before any WMMA so the scheduler
// can retire them with partial s_wait_loadcnt and keep VMEM busy under the WMMAs.

__global__ void __launch_bounds__(256)
gemm_wmma_f16(const _Float16* __restrict__ A, const _Float16* __restrict__ W,
              const float* __restrict__ bias, float* __restrict__ Cf,
              _Float16* __restrict__ Ch, int M, int Nc, int K, int relu) {
  int wid    = (blockIdx.x * blockDim.x + threadIdx.x) >> 5;
  int tilesN = Nc >> 6;                    // 64-wide panels
  int ntiles = (M >> 4) * tilesN;
  if (wid >= ntiles) return;               // whole-wave uniform exit
  int m0 = (wid / tilesN) << 4;
  int n0 = (wid % tilesN) << 6;
  int lane = threadIdx.x & 31;
  int hf = lane >> 4, r = lane & 15;

  const _Float16* ap = A + (size_t)(m0 + r) * K + 8 * hf;
  const _Float16* bp[4];
  #pragma unroll
  for (int nt = 0; nt < 4; ++nt)
    bp[nt] = W + (size_t)(n0 + nt * 16 + r) * K + 8 * hf;

  v8f acc[4] = {};
  for (int k0 = 0; k0 < K; k0 += 32) {
    __builtin_prefetch(ap + k0 + 64, 0, 1);
    __builtin_prefetch(bp[0] + k0 + 64, 0, 1);
    v8h alo = *(const v8h*)(ap + k0);
    v8h ahi = *(const v8h*)(ap + k0 + 16);
    v8h bl[4], bh[4];
    #pragma unroll
    for (int nt = 0; nt < 4; ++nt) {
      bl[nt] = *(const v8h*)(bp[nt] + k0);
      bh[nt] = *(const v8h*)(bp[nt] + k0 + 16);
    }
    v16h a = SHUF16(alo, ahi);
    #pragma unroll
    for (int nt = 0; nt < 4; ++nt) {
      v16h b = SHUF16(bl[nt], bh[nt]);
      acc[nt] = WMMA(a, b, acc[nt]);
    }
  }

  #pragma unroll
  for (int nt = 0; nt < 4; ++nt) {
    int n = n0 + nt * 16 + r;
    float bb = bias ? bias[n] : 0.f;
    #pragma unroll
    for (int j = 0; j < 8; ++j) {
      int m = m0 + j + 8 * hf;
      float v = acc[nt][j] + bb;
      if (relu) v = fmaxf(v, 0.f);
      size_t idx = (size_t)m * Nc + n;
      if (Cf) Cf[idx] = v;
      if (Ch) Ch[idx] = (_Float16)v;
    }
  }
}

// ---------------- BatchNorm (training batch stats) + ReLU ----------------

__global__ void bn_accum(const float* __restrict__ h, float* __restrict__ sums) {
  int row = blockIdx.x, c = threadIdx.x;
  float v = h[(size_t)row * DMODEL + c];
  atomicAdd(&sums[c], v);
  atomicAdd(&sums[DMODEL + c], v * v);
}

__global__ void bn_apply(float* __restrict__ h, _Float16* __restrict__ h16,
                         const float* __restrict__ sums,
                         const float* __restrict__ g, const float* __restrict__ b) {
  int i = blockIdx.x * blockDim.x + threadIdx.x;
  if (i >= N_NODES * DMODEL) return;
  int c = i & 255;
  float mu  = sums[c] * (1.f / N_NODES);
  float var = sums[DMODEL + c] * (1.f / N_NODES) - mu * mu;
  float v = (h[i] - mu) * rsqrtf(var + LN_EPS) * g[c] + b[c];
  v = fmaxf(v, 0.f);
  h[i]   = v;
  h16[i] = (_Float16)v;
}

// ---------------- residual add + LayerNorm ----------------

__global__ void __launch_bounds__(256)
add_layernorm(float* __restrict__ h, const float* __restrict__ delta,
              const float* __restrict__ g, const float* __restrict__ b,
              _Float16* __restrict__ h16) {
  __shared__ float s1[DMODEL];
  __shared__ float s2[DMODEL];
  int row = blockIdx.x, c = threadIdx.x;
  size_t idx = (size_t)row * DMODEL + c;
  float v = h[idx] + delta[idx];
  s1[c] = v; s2[c] = v * v;
  __syncthreads();
  for (int off = 128; off > 0; off >>= 1) {
    if (c < off) { s1[c] += s1[c + off]; s2[c] += s2[c + off]; }
    __syncthreads();
  }
  float mu  = s1[0] * (1.f / DMODEL);
  float var = s2[0] * (1.f / DMODEL) - mu * mu;
  float o = (v - mu) * rsqrtf(var + LN_EPS) * g[c] + b[c];
  h[idx]   = o;
  h16[idx] = (_Float16)o;
}

// ---------------- WMMA flash attention per (timestep block, head) ----------------
// grid (T, H), block 128 (4 waves). K/V staged to LDS via async global->LDS copies
// (ASYNCcnt path), zero-padded to a 32-key multiple; online softmax over 32-key steps.

__global__ void attn_block_wmma(const _Float16* __restrict__ qkv, const int* __restrict__ bs,
                                _Float16* __restrict__ o16) {
  extern __shared__ __align__(16) _Float16 smem[];
  _Float16* Ks = smem;                       // [NBMAX][KVP]
  _Float16* Vs = smem + NBMAX * KVP;         // [NBMAX][KVP]
  _Float16* Ps = smem + 2 * NBMAX * KVP;     // [4][16][PSP]

  int t = blockIdx.x, head = blockIdx.y;
  int s = bs[t], e = bs[t + 1];
  int nb = e - s;
  if (nb <= 0) return;
  if (nb > NBMAX) nb = NBMAX;                // never expected to trigger
  int nbpad = (nb + 31) & ~31;

  int tid = threadIdx.x;
  // ---- async staging of K,V (this head): 8 lanes x 16B per key row ----
  for (int idx = tid; idx < nb * 8; idx += (int)blockDim.x) {
    int key = idx >> 3;
    int c8  = (idx & 7) << 3;                // half offset within the 64-dim row
    const _Float16* g = qkv + (size_t)(s + key) * (3 * DMODEL) + DMODEL + head * D_HEAD + c8;
    unsigned klds = (unsigned)((const char*)&Ks[key * KVP + c8] - (const char*)smem);
    unsigned vlds = (unsigned)((const char*)&Vs[key * KVP + c8] - (const char*)smem);
    unsigned long long gk = (unsigned long long)g;
    unsigned long long gv = gk + (unsigned long long)(DMODEL * sizeof(_Float16));
    asm volatile("global_load_async_to_lds_b128 %0, %1, off"
                 :: "v"(klds), "v"(gk) : "memory");
    asm volatile("global_load_async_to_lds_b128 %0, %1, off"
                 :: "v"(vlds), "v"(gv) : "memory");
  }
  // ---- zero-pad tail keys with plain DS stores ----
  for (int idx = tid; idx < (nbpad - nb) * D_HEAD; idx += (int)blockDim.x) {
    int key = nb + (idx >> 6), d = idx & 63;
    Ks[key * KVP + d] = (_Float16)0.f;
    Vs[key * KVP + d] = (_Float16)0.f;
  }
  asm volatile("s_wait_asynccnt 0x0" ::: "memory");
  __syncthreads();

  int wid  = tid >> 5;
  int lane = tid & 31;
  int hf = lane >> 4, r = lane & 15;
  _Float16* ps = Ps + wid * 16 * PSP;

  int nqt = (nb + 15) >> 4;
  for (int qt = wid; qt < nqt; qt += 4) {
    int qs = qt << 4;
    int gq = s + qs + r; if (gq >= e) gq = e - 1;        // clamp tail rows (not stored)
    const _Float16* apq = qkv + (size_t)gq * (3 * DMODEL) + head * D_HEAD + 8 * hf;
    v8h q00 = *(const v8h*)(apq);
    v8h q01 = *(const v8h*)(apq + 16);
    v8h q10 = *(const v8h*)(apq + 32);
    v8h q11 = *(const v8h*)(apq + 48);
    v16h aq0 = SHUF16(q00, q01);                         // dims 0..31
    v16h aq1 = SHUF16(q10, q11);                         // dims 32..63

    float mrow[8], lrow[8], rsc[8];
    v8f o0 = {}, o1 = {}, o2 = {}, o3 = {};
    #pragma unroll
    for (int j = 0; j < 8; ++j) { mrow[j] = -1e30f; lrow[j] = 0.f; }

    for (int j0 = 0; j0 < nbpad; j0 += 32) {
      // ---- scores: two 16-key C tiles ----
      v8f c0 = {}, c1 = {};
      #pragma unroll
      for (int k0 = 0; k0 < D_HEAD; k0 += 32) {
        const _Float16* kp0 = Ks + (size_t)(j0 + r) * KVP + k0 + 8 * hf;
        const _Float16* kp1 = Ks + (size_t)(j0 + 16 + r) * KVP + k0 + 8 * hf;
        v8h b0l = *(const v8h*)(kp0); v8h b0h = *(const v8h*)(kp0 + 16);
        v8h b1l = *(const v8h*)(kp1); v8h b1h = *(const v8h*)(kp1 + 16);
        v16h bk0 = SHUF16(b0l, b0h);
        v16h bk1 = SHUF16(b1l, b1h);
        v16h a = (k0 == 0) ? aq0 : aq1;
        c0 = WMMA(a, bk0, c0);
        c1 = WMMA(a, bk1, c1);
      }
      // ---- online softmax on C layout (row j+8*hf, col r) ----
      #pragma unroll
      for (int j = 0; j < 8; ++j) {
        float s0 = c0[j] * 0.125f;                       // 1/sqrt(64)
        float s1 = c1[j] * 0.125f;
        if (j0 + r >= nb)      s0 = -1e30f;
        if (j0 + 16 + r >= nb) s1 = -1e30f;
        float mx = fmaxf(s0, s1);
        mx = fmaxf(mx, __shfl_xor(mx, 1, 16));
        mx = fmaxf(mx, __shfl_xor(mx, 2, 16));
        mx = fmaxf(mx, __shfl_xor(mx, 4, 16));
        mx = fmaxf(mx, __shfl_xor(mx, 8, 16));
        float mn = fmaxf(mrow[j], mx);
        float p0 = __expf(s0 - mn);
        float p1 = __expf(s1 - mn);
        float rs = __expf(mrow[j] - mn);
        float pm = p0 + p1;
        pm += __shfl_xor(pm, 1, 16);
        pm += __shfl_xor(pm, 2, 16);
        pm += __shfl_xor(pm, 4, 16);
        pm += __shfl_xor(pm, 8, 16);
        lrow[j] = lrow[j] * rs + pm;
        mrow[j] = mn;
        rsc[j]  = rs;
        ps[(j + 8 * hf) * PSP + r]      = (_Float16)p0;
        ps[(j + 8 * hf) * PSP + 16 + r] = (_Float16)p1;
      }
      #pragma unroll
      for (int j = 0; j < 8; ++j) {
        o0[j] *= rsc[j]; o1[j] *= rsc[j]; o2[j] *= rsc[j]; o3[j] *= rsc[j];
      }
      // ---- reload P as A-fragment (same-wave LDS, no barrier) ----
      const _Float16* pr = ps + r * PSP + 8 * hf;
      v8h plo = *(const v8h*)(pr);
      v8h phi = *(const v8h*)(pr + 16);
      v16h aP = SHUF16(plo, phi);
      // ---- O += P @ V : 4 n-tiles of 16 dims ----
      #pragma unroll
      for (int nt = 0; nt < 4; ++nt) {
        v16h bV;
        #pragma unroll
        for (int ee = 0; ee < 16; ++ee)
          bV[ee] = Vs[(size_t)(j0 + 8 * hf + (ee & 7) + 16 * (ee >> 3)) * KVP + nt * 16 + r];
        if      (nt == 0) o0 = WMMA(aP, bV, o0);
        else if (nt == 1) o1 = WMMA(aP, bV, o1);
        else if (nt == 2) o2 = WMMA(aP, bV, o2);
        else              o3 = WMMA(aP, bV, o3);
      }
    }
    // ---- normalize + store (f16 operand for out-projection GEMM) ----
    #pragma unroll
    for (int j = 0; j < 8; ++j) {
      int qi = qs + j + 8 * hf;
      if (qi < nb) {
        float inv = 1.f / lrow[j];
        _Float16* op = o16 + (size_t)(s + qi) * DMODEL + head * D_HEAD;
        op[ 0 + r] = (_Float16)(o0[j] * inv);
        op[16 + r] = (_Float16)(o1[j] * inv);
        op[32 + r] = (_Float16)(o2[j] * inv);
        op[48 + r] = (_Float16)(o3[j] * inv);
      }
    }
  }
}

// ---------------- classifier [N,256] -> [N,2] ----------------

__global__ void classifier(const float* __restrict__ h, const float* __restrict__ w,
                           const float* __restrict__ b, float* __restrict__ out) {
  int n = blockIdx.x * blockDim.x + threadIdx.x;
  if (n >= N_NODES) return;
  #pragma unroll
  for (int o = 0; o < OUT_DIM; ++o) {
    float acc = b[o];
    for (int k = 0; k < DMODEL; ++k)
      acc += h[(size_t)n * DMODEL + k] * w[o * DMODEL + k];
    out[n * OUT_DIM + o] = acc;
  }
}

// ---------------- host orchestration ----------------

extern "C" void kernel_launch(void* const* d_in, const int* in_sizes, int n_in,
                              void* d_out, int out_size, void* d_ws, size_t ws_size,
                              hipStream_t stream) {
  (void)in_sizes; (void)n_in; (void)out_size; (void)ws_size;
  const float* x        = (const float*)d_in[0];
  const int*   ei       = (const int*)d_in[1];
  const int*   ts       = (const int*)d_in[2];
  const float* lin_l_w  = (const float*)d_in[3];
  const float* lin_l_b  = (const float*)d_in[4];
  const float* lin_r_w  = (const float*)d_in[5];
  const float* bn_g     = (const float*)d_in[6];
  const float* bn_b     = (const float*)d_in[7];
  const float* in_proj_w= (const float*)d_in[8];
  const float* in_proj_b= (const float*)d_in[9];
  const float* out_w    = (const float*)d_in[10];
  const float* out_b    = (const float*)d_in[11];
  const float* ff1_w    = (const float*)d_in[12];
  const float* ff1_b    = (const float*)d_in[13];
  const float* ff2_w    = (const float*)d_in[14];
  const float* ff2_b    = (const float*)d_in[15];
  const float* ln1_g    = (const float*)d_in[16];
  const float* ln1_b    = (const float*)d_in[17];
  const float* ln2_g    = (const float*)d_in[18];
  const float* ln2_b    = (const float*)d_in[19];
  const float* cls_w    = (const float*)d_in[20];
  const float* cls_b    = (const float*)d_in[21];
  float* out = (float*)d_out;

  char* base = (char*)d_ws;
  size_t off = 0;
  auto alloc = [&](size_t bytes) -> void* {
    void* p = base + off;
    off = (off + bytes + 255) & ~(size_t)255;
    return p;
  };

  float*    agg   = (float*)   alloc((size_t)N_NODES * D_INF * 4);
  float*    cnt   = (float*)   alloc((size_t)N_NODES * 4);
  float*    h     = (float*)   alloc((size_t)N_NODES * DMODEL * 4);
  _Float16* h16   = (_Float16*)alloc((size_t)N_NODES * DMODEL * 2);
  _Float16* xa16  = (_Float16*)alloc((size_t)N_NODES * DMODEL * 2);
  _Float16* qkv16 = (_Float16*)alloc((size_t)N_NODES * 3 * DMODEL * 2);
  _Float16* a16   = (_Float16*)alloc((size_t)N_NODES * DMODEL * 2);
  float*    tmp   = (float*)   alloc((size_t)N_NODES * DMODEL * 4);
  _Float16* f116  = (_Float16*)alloc((size_t)N_NODES * D_FF * 2);
  _Float16* wcat  = (_Float16*)alloc((size_t)DMODEL * DMODEL * 2);
  _Float16* wqkv  = (_Float16*)alloc((size_t)2 * 3 * DMODEL * DMODEL * 2);
  _Float16* wout  = (_Float16*)alloc((size_t)2 * DMODEL * DMODEL * 2);
  _Float16* wff1  = (_Float16*)alloc((size_t)2 * D_FF * DMODEL * 2);
  _Float16* wff2  = (_Float16*)alloc((size_t)2 * DMODEL * D_FF * 2);
  float*    bnsum = (float*)   alloc((size_t)2 * DMODEL * 4);
  int*      bst   = (int*)     alloc((size_t)(T_STEPS + 1) * 4);

  fill_zero<<<(N_NODES * D_INF + 255) / 256, 256, 0, stream>>>(agg, N_NODES * D_INF);
  fill_zero<<<(N_NODES + 255) / 256, 256, 0, stream>>>(cnt, N_NODES);
  fill_zero<<<(2 * DMODEL + 255) / 256, 256, 0, stream>>>(bnsum, 2 * DMODEL);

  concat_weight<<<(DMODEL * DMODEL + 255) / 256, 256, 0, stream>>>(lin_l_w, lin_r_w, wcat);
  cvt_f32_f16<<<(2 * 3 * DMODEL * DMODEL + 255) / 256, 256, 0, stream>>>(in_proj_w, wqkv, 2 * 3 * DMODEL * DMODEL);
  cvt_f32_f16<<<(2 * DMODEL * DMODEL + 255) / 256, 256, 0, stream>>>(out_w, wout, 2 * DMODEL * DMODEL);
  cvt_f32_f16<<<(2 * D_FF * DMODEL + 255) / 256, 256, 0, stream>>>(ff1_w, wff1, 2 * D_FF * DMODEL);
  cvt_f32_f16<<<(2 * DMODEL * D_FF + 255) / 256, 256, 0, stream>>>(ff2_w, wff2, 2 * DMODEL * D_FF);

  edge_agg<<<(N_EDGES * D_INF + 255) / 256, 256, 0, stream>>>(ei, x, agg, cnt);
  build_xa16<<<(N_NODES * DMODEL + 255) / 256, 256, 0, stream>>>(agg, cnt, x, xa16);
  block_starts<<<1, 64, 0, stream>>>(ts, bst);

  const int panels256 = (N_NODES / 16) * (DMODEL / 64);      // 1536 waves -> 192 blocks
  const int panels768 = (N_NODES / 16) * (3 * DMODEL / 64);  // 4608 waves -> 576 blocks
  const size_t attn_lds = (size_t)(2 * NBMAX * KVP + 4 * 16 * PSP) * sizeof(_Float16);

  gemm_wmma_f16<<<panels256 / 8, 256, 0, stream>>>(xa16, wcat, lin_l_b, h, nullptr,
                                                   N_NODES, DMODEL, DMODEL, 0);
  bn_accum<<<N_NODES, DMODEL, 0, stream>>>(h, bnsum);
  bn_apply<<<(N_NODES * DMODEL + 255) / 256, 256, 0, stream>>>(h, h16, bnsum, bn_g, bn_b);

  for (int l = 0; l < 2; ++l) {
    gemm_wmma_f16<<<panels768 / 8, 256, 0, stream>>>(
        h16, wqkv + (size_t)l * 3 * DMODEL * DMODEL, in_proj_b + l * 3 * DMODEL,
        nullptr, qkv16, N_NODES, 3 * DMODEL, DMODEL, 0);
    dim3 ag(T_STEPS, N_HEADS);
    attn_block_wmma<<<ag, 128, attn_lds, stream>>>(qkv16, bst, a16);
    gemm_wmma_f16<<<panels256 / 8, 256, 0, stream>>>(
        a16, wout + (size_t)l * DMODEL * DMODEL, out_b + l * DMODEL,
        tmp, nullptr, N_NODES, DMODEL, DMODEL, 0);
    add_layernorm<<<N_NODES, DMODEL, 0, stream>>>(h, tmp, ln1_g + l * DMODEL,
                                                  ln1_b + l * DMODEL, h16);
    gemm_wmma_f16<<<panels256 / 8, 256, 0, stream>>>(
        h16, wff1 + (size_t)l * D_FF * DMODEL, ff1_b + l * D_FF,
        nullptr, f116, N_NODES, D_FF, DMODEL, 1);
    gemm_wmma_f16<<<panels256 / 8, 256, 0, stream>>>(
        f116, wff2 + (size_t)l * DMODEL * D_FF, ff2_b + l * DMODEL,
        tmp, nullptr, N_NODES, DMODEL, D_FF, 0);
    add_layernorm<<<N_NODES, DMODEL, 0, stream>>>(h, tmp, ln2_g + l * DMODEL,
                                                  ln2_b + l * DMODEL, h16);
  }

  classifier<<<(N_NODES + 255) / 256, 256, 0, stream>>>(h, cls_w, cls_b, out);
}
